// EquivariantBlock_45973329936456
// MI455X (gfx1250) — compile-verified, matrix operand order checked by reading, not run on hardware
//
#include <hip/hip_runtime.h>
#include <cstddef>

#define HH 256
#define MAXKP 800   // pad of 770 (3H+2) to multiple of 32
#define KEQN 544    // pad of 514 (2H+2)

typedef __attribute__((ext_vector_type(16))) __bf16 v16bf;
typedef __attribute__((ext_vector_type(8)))  float  v8f;

// Fast activations: v_exp_f32 (TRANS32, co-issues with WMMA) + v_rcp_f32.
__device__ __forceinline__ float fast_exp(float x) { return __expf(x); }
__device__ __forceinline__ float fast_rcp(float x) { return __builtin_amdgcn_rcpf(x); }
__device__ __forceinline__ float siluf(float x) { return x * fast_rcp(1.f + fast_exp(-x)); }
__device__ __forceinline__ float sigmf(float x) { return fast_rcp(1.f + fast_exp(-x)); }

// ---------------------------------------------------------------------------
// Fused per-edge MLP:
//   x = concat(h[row], h[col], eattr2[, bondEmb[etype]])
//   m1  = silu(x @ W1^T + b1)           (WMMA bf16, f32 acc)
//   mij = silu(m1 @ W2^T + b2)          (WMMA)
// MODE 0 (GCL): gate = sigmoid(mij . wlast + blast); scatter mij*gate*scale
//               into agg[row] (H cols, atomic f32)
// MODE 1 (EQ):  s = mij . wlast; scatter cdiff[e]*s*scale into agg[row] (3 cols)
// ---------------------------------------------------------------------------
template <int MODE>
__global__ __launch_bounds__(256)
void edge_mlp_kernel(const float* __restrict__ h,
                     const int* __restrict__ row, const int* __restrict__ col,
                     const float* __restrict__ eattr2,
                     const int* __restrict__ etype, const float* __restrict__ bondEmb,
                     const float* __restrict__ W1, const float* __restrict__ b1,
                     const float* __restrict__ W2, const float* __restrict__ b2,
                     const float* __restrict__ wlast, const float* __restrict__ blast,
                     const float* __restrict__ cdiff,
                     float* __restrict__ aggOut,
                     int E, int Kin, float outScale)
{
    __shared__ __bf16 X[16][MAXKP];
    __shared__ __bf16 M1[16][HH];
    __shared__ float  MIJ[16][HH];
    __shared__ float  dotp[16][16];
    __shared__ float  gate[16];
    __shared__ int    erow[16];

    const int tid = threadIdx.x;
    const int e0  = blockIdx.x * 16;
    const int Kp  = (Kin + 31) & ~31;

    if (tid < 16) {
        int e = e0 + tid;
        erow[tid] = (e < E) ? row[e] : 0;
    }

    // Gather + f32->bf16 convert into LDS (zero padding beyond Kin / E)
    for (int idx = tid; idx < 16 * Kp; idx += 256) {
        int r = idx / Kp;
        int k = idx - r * Kp;
        int e = e0 + r;
        float v = 0.f;
        if (e < E && k < Kin) {
            if (k < HH)              v = h[(size_t)row[e] * HH + k];
            else if (k < 2 * HH)     v = h[(size_t)col[e] * HH + (k - HH)];
            else if (k < 2 * HH + 2) v = eattr2[(size_t)e * 2 + (k - 2 * HH)];
            else                     v = bondEmb[(size_t)etype[e] * HH + (k - 2 * HH - 2)];
        }
        X[r][k] = (__bf16)v;
    }
    __syncthreads();

    const int lane = tid & 31;
    const int wv   = tid >> 5;      // 8 waves -> each owns 32 output cols
    const int m    = lane & 15;
    const int hi   = lane >> 4;

    // ---- GEMM1: [16 x Kin] @ W1^T -> [16 x 256] ----
    v8f acc0 = {}; v8f acc1 = {};
    for (int k0 = 0; k0 < Kp; k0 += 32) {
        v16bf a;
#pragma unroll
        for (int j = 0; j < 16; ++j) {
            int r  = j >> 1;
            int kk = ((r >= 4) ? 16 : 0) + (hi ? 8 : 0) + ((r & 3) << 1) + (j & 1);
            a[j] = X[m][k0 + kk];
        }
#pragma unroll
        for (int t = 0; t < 2; ++t) {
            int n = wv * 32 + t * 16 + m;
            v16bf b;
#pragma unroll
            for (int j = 0; j < 16; ++j) {
                int r  = j >> 1;
                int kk = ((r >= 4) ? 16 : 0) + (hi ? 8 : 0) + ((r & 3) << 1) + (j & 1);
                int k  = k0 + kk;
                b[j] = (k < Kin) ? (__bf16)W1[(size_t)n * Kin + k] : (__bf16)0.f;
            }
            if (t == 0)
                acc0 = __builtin_amdgcn_wmma_f32_16x16x32_bf16(false, a, false, b, (short)0, acc0, false, false);
            else
                acc1 = __builtin_amdgcn_wmma_f32_16x16x32_bf16(false, a, false, b, (short)0, acc1, false, false);
        }
    }
#pragma unroll
    for (int t = 0; t < 2; ++t) {
        int n = wv * 32 + t * 16 + m;
        float bb = b1[n];
        const v8f& ac = t ? acc1 : acc0;
#pragma unroll
        for (int i = 0; i < 8; ++i)
            M1[i + 8 * hi][n] = (__bf16)siluf(ac[i] + bb);
    }
    __syncthreads();

    // ---- GEMM2: [16 x 256] @ W2^T -> [16 x 256] ----
    v8f c0 = {}; v8f c1 = {};
    for (int k0 = 0; k0 < HH; k0 += 32) {
        v16bf a;
#pragma unroll
        for (int j = 0; j < 16; ++j) {
            int r  = j >> 1;
            int kk = ((r >= 4) ? 16 : 0) + (hi ? 8 : 0) + ((r & 3) << 1) + (j & 1);
            a[j] = M1[m][k0 + kk];
        }
#pragma unroll
        for (int t = 0; t < 2; ++t) {
            int n = wv * 32 + t * 16 + m;
            v16bf b;
#pragma unroll
            for (int j = 0; j < 16; ++j) {
                int r  = j >> 1;
                int kk = ((r >= 4) ? 16 : 0) + (hi ? 8 : 0) + ((r & 3) << 1) + (j & 1);
                b[j] = (__bf16)W2[(size_t)n * HH + (k0 + kk)];
            }
            if (t == 0)
                c0 = __builtin_amdgcn_wmma_f32_16x16x32_bf16(false, a, false, b, (short)0, c0, false, false);
            else
                c1 = __builtin_amdgcn_wmma_f32_16x16x32_bf16(false, a, false, b, (short)0, c1, false, false);
        }
    }
#pragma unroll
    for (int t = 0; t < 2; ++t) {
        int n = wv * 32 + t * 16 + m;
        float bb = b2[n];
        const v8f& ac = t ? c1 : c0;
#pragma unroll
        for (int i = 0; i < 8; ++i)
            MIJ[i + 8 * hi][n] = siluf(ac[i] + bb);
    }
    __syncthreads();

    // ---- last linear (H -> 1): per-edge dot with wlast ----
    {
        int e = tid >> 4, c = tid & 15;
        float p = 0.f;
#pragma unroll
        for (int j = 0; j < 16; ++j) p += MIJ[e][c * 16 + j] * wlast[c * 16 + j];
        dotp[e][c] = p;
    }
    __syncthreads();
    if (tid < 16) {
        float s = 0.f;
#pragma unroll
        for (int c = 0; c < 16; ++c) s += dotp[tid][c];
        if (MODE == 0) gate[tid] = sigmf(s + blast[0]);
        else           gate[tid] = s;
    }
    __syncthreads();

    if (MODE == 0) {
        int e = tid >> 4, c16 = tid & 15;
        if (e0 + e < E) {
            float g = gate[e] * outScale;
            int rdst = erow[e];
#pragma unroll
            for (int j = 0; j < 16; ++j) {
                int cc = j * 16 + c16;
                atomicAdd(&aggOut[(size_t)rdst * HH + cc], MIJ[e][cc] * g);
            }
        }
    } else {
        if (tid < 48) {
            int e = tid / 3, d = tid - e * 3;
            if (e0 + e < E)
                atomicAdd(&aggOut[(size_t)erow[e] * 3 + d],
                          cdiff[(size_t)(e0 + e) * 3 + d] * gate[e] * outScale);
        }
    }
}

// ---------------------------------------------------------------------------
// Fused node update: hout = h + ( silu([h|aux|agg] @ W1^T + b1) @ W2^T + b2 )
// ---------------------------------------------------------------------------
__global__ __launch_bounds__(256)
void node_gcl_kernel(const float* __restrict__ h, const float* __restrict__ aux,
                     const float* __restrict__ agg,
                     const float* __restrict__ W1, const float* __restrict__ b1,
                     const float* __restrict__ W2, const float* __restrict__ b2,
                     float* __restrict__ hout, int N)
{
    __shared__ __bf16 X[16][3 * HH];
    __shared__ __bf16 M1[16][HH];
    const int tid = threadIdx.x;
    const int n0  = blockIdx.x * 16;

    for (int idx = tid; idx < 16 * 3 * HH; idx += 256) {
        int r = idx / (3 * HH), k = idx - r * (3 * HH);
        int nd = n0 + r;
        float v = 0.f;
        if (nd < N) {
            if (k < HH)          v = h[(size_t)nd * HH + k];
            else if (k < 2 * HH) v = aux[(size_t)nd * HH + (k - HH)];
            else                 v = agg[(size_t)nd * HH + (k - 2 * HH)];
        }
        X[r][k] = (__bf16)v;
    }
    __syncthreads();

    const int lane = tid & 31, wv = tid >> 5, m = lane & 15, hi = lane >> 4;

    v8f a0 = {}; v8f a1 = {};
    for (int k0 = 0; k0 < 3 * HH; k0 += 32) {
        v16bf a;
#pragma unroll
        for (int j = 0; j < 16; ++j) {
            int r  = j >> 1;
            int kk = ((r >= 4) ? 16 : 0) + (hi ? 8 : 0) + ((r & 3) << 1) + (j & 1);
            a[j] = X[m][k0 + kk];
        }
#pragma unroll
        for (int t = 0; t < 2; ++t) {
            int n = wv * 32 + t * 16 + m;
            v16bf b;
#pragma unroll
            for (int j = 0; j < 16; ++j) {
                int r  = j >> 1;
                int kk = ((r >= 4) ? 16 : 0) + (hi ? 8 : 0) + ((r & 3) << 1) + (j & 1);
                b[j] = (__bf16)W1[(size_t)n * (3 * HH) + (k0 + kk)];
            }
            if (t == 0)
                a0 = __builtin_amdgcn_wmma_f32_16x16x32_bf16(false, a, false, b, (short)0, a0, false, false);
            else
                a1 = __builtin_amdgcn_wmma_f32_16x16x32_bf16(false, a, false, b, (short)0, a1, false, false);
        }
    }
#pragma unroll
    for (int t = 0; t < 2; ++t) {
        int n = wv * 32 + t * 16 + m;
        float bb = b1[n];
        const v8f& ac = t ? a1 : a0;
#pragma unroll
        for (int i = 0; i < 8; ++i)
            M1[i + 8 * hi][n] = (__bf16)siluf(ac[i] + bb);
    }
    __syncthreads();

    v8f c0 = {}; v8f c1 = {};
    for (int k0 = 0; k0 < HH; k0 += 32) {
        v16bf a;
#pragma unroll
        for (int j = 0; j < 16; ++j) {
            int r  = j >> 1;
            int kk = ((r >= 4) ? 16 : 0) + (hi ? 8 : 0) + ((r & 3) << 1) + (j & 1);
            a[j] = M1[m][k0 + kk];
        }
#pragma unroll
        for (int t = 0; t < 2; ++t) {
            int n = wv * 32 + t * 16 + m;
            v16bf b;
#pragma unroll
            for (int j = 0; j < 16; ++j) {
                int r  = j >> 1;
                int kk = ((r >= 4) ? 16 : 0) + (hi ? 8 : 0) + ((r & 3) << 1) + (j & 1);
                b[j] = (__bf16)W2[(size_t)n * HH + (k0 + kk)];
            }
            if (t == 0)
                c0 = __builtin_amdgcn_wmma_f32_16x16x32_bf16(false, a, false, b, (short)0, c0, false, false);
            else
                c1 = __builtin_amdgcn_wmma_f32_16x16x32_bf16(false, a, false, b, (short)0, c1, false, false);
        }
    }
#pragma unroll
    for (int t = 0; t < 2; ++t) {
        int n = wv * 32 + t * 16 + m;
        float bb = b2[n];
        const v8f& ac = t ? c1 : c0;
#pragma unroll
        for (int i = 0; i < 8; ++i) {
            int nd = n0 + i + 8 * hi;
            if (nd < N)
                hout[(size_t)nd * HH + n] = h[(size_t)nd * HH + n] + ac[i] + bb;
        }
    }
}

// ---------------------------------------------------------------------------
// Equivariant fragment branch (per-node):
//   s = silu(silu([h_a|bmh|r|r] @ W1^T + b1) @ W2^T + b2) . w3
//   coordOut = x_a + coordAgg + cdf * s      (coordAgg pre-scaled by 1/100)
// ---------------------------------------------------------------------------
__global__ __launch_bounds__(256)
void node_eq_kernel(const float* __restrict__ ha, const float* __restrict__ bmh,
                    const float* __restrict__ radf,
                    const float* __restrict__ W1, const float* __restrict__ b1,
                    const float* __restrict__ W2, const float* __restrict__ b2,
                    const float* __restrict__ w3,
                    const float* __restrict__ cdf, const float* __restrict__ xa,
                    const float* __restrict__ coordAgg, float* __restrict__ coordOut,
                    int N)
{
    __shared__ __bf16 X[16][KEQN];
    __shared__ __bf16 M1[16][HH];
    __shared__ float  MIJ[16][HH];
    __shared__ float  dotp[16][16];
    __shared__ float  sgl[16];
    const int Kin = 2 * HH + 2;   // 514
    const int tid = threadIdx.x;
    const int n0  = blockIdx.x * 16;

    for (int idx = tid; idx < 16 * KEQN; idx += 256) {
        int r = idx / KEQN, k = idx - r * KEQN;
        int nd = n0 + r;
        float v = 0.f;
        if (nd < N && k < Kin) {
            if (k < HH)          v = ha[(size_t)nd * HH + k];
            else if (k < 2 * HH) v = bmh[(size_t)nd * HH + (k - HH)];
            else                 v = radf[nd];
        }
        X[r][k] = (__bf16)v;
    }
    __syncthreads();

    const int lane = tid & 31, wv = tid >> 5, m = lane & 15, hi = lane >> 4;

    v8f a0 = {}; v8f a1 = {};
    for (int k0 = 0; k0 < KEQN; k0 += 32) {
        v16bf a;
#pragma unroll
        for (int j = 0; j < 16; ++j) {
            int r  = j >> 1;
            int kk = ((r >= 4) ? 16 : 0) + (hi ? 8 : 0) + ((r & 3) << 1) + (j & 1);
            a[j] = X[m][k0 + kk];
        }
#pragma unroll
        for (int t = 0; t < 2; ++t) {
            int n = wv * 32 + t * 16 + m;
            v16bf b;
#pragma unroll
            for (int j = 0; j < 16; ++j) {
                int r  = j >> 1;
                int kk = ((r >= 4) ? 16 : 0) + (hi ? 8 : 0) + ((r & 3) << 1) + (j & 1);
                int k  = k0 + kk;
                b[j] = (k < Kin) ? (__bf16)W1[(size_t)n * Kin + k] : (__bf16)0.f;
            }
            if (t == 0)
                a0 = __builtin_amdgcn_wmma_f32_16x16x32_bf16(false, a, false, b, (short)0, a0, false, false);
            else
                a1 = __builtin_amdgcn_wmma_f32_16x16x32_bf16(false, a, false, b, (short)0, a1, false, false);
        }
    }
#pragma unroll
    for (int t = 0; t < 2; ++t) {
        int n = wv * 32 + t * 16 + m;
        float bb = b1[n];
        const v8f& ac = t ? a1 : a0;
#pragma unroll
        for (int i = 0; i < 8; ++i)
            M1[i + 8 * hi][n] = (__bf16)siluf(ac[i] + bb);
    }
    __syncthreads();

    v8f c0 = {}; v8f c1 = {};
    for (int k0 = 0; k0 < HH; k0 += 32) {
        v16bf a;
#pragma unroll
        for (int j = 0; j < 16; ++j) {
            int r  = j >> 1;
            int kk = ((r >= 4) ? 16 : 0) + (hi ? 8 : 0) + ((r & 3) << 1) + (j & 1);
            a[j] = M1[m][k0 + kk];
        }
#pragma unroll
        for (int t = 0; t < 2; ++t) {
            int n = wv * 32 + t * 16 + m;
            v16bf b;
#pragma unroll
            for (int j = 0; j < 16; ++j) {
                int r  = j >> 1;
                int kk = ((r >= 4) ? 16 : 0) + (hi ? 8 : 0) + ((r & 3) << 1) + (j & 1);
                b[j] = (__bf16)W2[(size_t)n * HH + (k0 + kk)];
            }
            if (t == 0)
                c0 = __builtin_amdgcn_wmma_f32_16x16x32_bf16(false, a, false, b, (short)0, c0, false, false);
            else
                c1 = __builtin_amdgcn_wmma_f32_16x16x32_bf16(false, a, false, b, (short)0, c1, false, false);
        }
    }
#pragma unroll
    for (int t = 0; t < 2; ++t) {
        int n = wv * 32 + t * 16 + m;
        float bb = b2[n];
        const v8f& ac = t ? c1 : c0;
#pragma unroll
        for (int i = 0; i < 8; ++i)
            MIJ[i + 8 * hi][n] = siluf(ac[i] + bb);
    }
    __syncthreads();

    {
        int e = tid >> 4, c = tid & 15;
        float p = 0.f;
#pragma unroll
        for (int j = 0; j < 16; ++j) p += MIJ[e][c * 16 + j] * w3[c * 16 + j];
        dotp[e][c] = p;
    }
    __syncthreads();
    if (tid < 16) {
        float s = 0.f;
#pragma unroll
        for (int c = 0; c < 16; ++c) s += dotp[tid][c];
        sgl[tid] = s;
    }
    __syncthreads();
    if (tid < 48) {
        int r = tid / 3, d = tid - r * 3;
        int nd = n0 + r;
        if (nd < N)
            coordOut[(size_t)nd * 3 + d] = xa[(size_t)nd * 3 + d]
                                         + coordAgg[(size_t)nd * 3 + d]
                                         + cdf[(size_t)nd * 3 + d] * sgl[r];
    }
}

// ---------------------------------------------------------------------------
// Sparse (0/1) pooling matmul: out[r,:] = sum_a M[r,a] * X[a,:]
// ---------------------------------------------------------------------------
__global__ __launch_bounds__(256)
void spmm_kernel(const float* __restrict__ M, const float* __restrict__ X,
                 float* __restrict__ out, int inner, int width, int binarize)
{
    __shared__ int   idxs[256];
    __shared__ float vals[256];
    __shared__ int   cnt;
    const int r = blockIdx.x, tid = threadIdx.x;
    float acc = 0.f;
    for (int a0 = 0; a0 < inner; a0 += 256) {
        if (tid == 0) cnt = 0;
        __syncthreads();
        int a = a0 + tid;
        if (a < inner) {
            float v = M[(size_t)r * inner + a];
            if (v != 0.f) {
                int p = atomicAdd(&cnt, 1);
                idxs[p] = a;
                vals[p] = binarize ? 1.f : v;
            }
        }
        __syncthreads();
        int c = cnt;
        if (tid < width)
            for (int i = 0; i < c; ++i)
                acc += vals[i] * X[(size_t)idxs[i] * width + tid];
        __syncthreads();
    }
    if (tid < width) out[(size_t)r * width + tid] = acc;
}

__global__ void c2d_kernel(const float* __restrict__ x, const int* __restrict__ row,
                           const int* __restrict__ col, const float* __restrict__ eattr,
                           float* __restrict__ attr2, float* __restrict__ cdiff, int E)
{
    int e = blockIdx.x * blockDim.x + threadIdx.x;
    if (e >= E) return;
    int r = row[e], c = col[e];
    float dx = x[(size_t)r * 3 + 0] - x[(size_t)c * 3 + 0];
    float dy = x[(size_t)r * 3 + 1] - x[(size_t)c * 3 + 1];
    float dz = x[(size_t)r * 3 + 2] - x[(size_t)c * 3 + 2];
    float rad = dx * dx + dy * dy + dz * dz;
    attr2[(size_t)e * 2 + 0] = rad;
    attr2[(size_t)e * 2 + 1] = eattr[e];
    if (cdiff) {
        float inv = fast_rcp(sqrtf(rad + 1e-8f) + 1.f);
        cdiff[(size_t)e * 3 + 0] = dx * inv;
        cdiff[(size_t)e * 3 + 1] = dy * inv;
        cdiff[(size_t)e * 3 + 2] = dz * inv;
    }
}

__global__ void cdf_kernel(const float* __restrict__ xa, const float* __restrict__ bmxf,
                           float* __restrict__ cdf, float* __restrict__ radf, int N)
{
    int i = blockIdx.x * blockDim.x + threadIdx.x;
    if (i >= N) return;
    float dx = xa[(size_t)i * 3 + 0] - bmxf[(size_t)i * 3 + 0];
    float dy = xa[(size_t)i * 3 + 1] - bmxf[(size_t)i * 3 + 1];
    float dz = xa[(size_t)i * 3 + 2] - bmxf[(size_t)i * 3 + 2];
    float rad = dx * dx + dy * dy + dz * dz;
    radf[i] = rad;
    float inv = fast_rcp(sqrtf(rad + 1e-8f) + 1.f);
    cdf[(size_t)i * 3 + 0] = dx * inv;
    cdf[(size_t)i * 3 + 1] = dy * inv;
    cdf[(size_t)i * 3 + 2] = dz * inv;
}

__global__ void fill_kernel(float* __restrict__ p, size_t n)
{
    size_t i = (size_t)blockIdx.x * blockDim.x + threadIdx.x;
    if (i < n) p[i] = 0.f;
}

// ---------------------------------------------------------------------------
extern "C" void kernel_launch(void* const* d_in, const int* in_sizes, int n_in,
                              void* d_out, int out_size, void* d_ws, size_t ws_size,
                              hipStream_t stream)
{
    // setup_inputs() insertion order:
    const float* h_a_in  = (const float*)d_in[0];
    const float* x_a     = (const float*)d_in[1];
    const int*   ea_idx  = (const int*)d_in[2];
    const int*   ea_type = (const int*)d_in[3];
    const float* ea_attr = (const float*)d_in[4];
    const float* h_f_in  = (const float*)d_in[5];
    const float* x_f     = (const float*)d_in[6];
    const int*   ef_idx  = (const int*)d_in[7];
    const float* ef_attr = (const float*)d_in[8];
    const float* m_mat   = (const float*)d_in[9];
    const float* bm_mat  = (const float*)d_in[10];

    const int NA = in_sizes[0] / HH;
    const int NF = in_sizes[5] / HH;
    const int EA = in_sizes[2] / 2;
    const int EF = in_sizes[7] / 2;
    const int* ar = ea_idx;       const int* ac = ea_idx + EA;
    const int* fr = ef_idx;       const int* fc = ef_idx + EF;

    // params pytree (jax sorted-key flatten): a_eq, a_inv_0, a_inv_1, f_inv_0, f_inv_1
    int pi = 11;
    auto P = [&]() { return (const float*)d_in[pi++]; };
    const float* eq_atom1_b = P(); const float* eq_atom1_w = P();
    const float* eq_atom2_b = P(); const float* eq_atom2_w = P();
    const float* eq_atom3_w = P();
    const float* eq_bond    = P();
    const float* eq_frag1_b = P(); const float* eq_frag1_w = P();
    const float* eq_frag2_b = P(); const float* eq_frag2_w = P();
    const float* eq_frag3_w = P();
    struct Gcl { const float *attb,*attw,*bond,*e1b,*e1w,*e2b,*e2w,*n1b,*n1w,*n2b,*n2w; };
    auto getGcl = [&]() {
        Gcl g;
        g.attb = P(); g.attw = P(); g.bond = P();
        g.e1b = P(); g.e1w = P(); g.e2b = P(); g.e2w = P();
        g.n1b = P(); g.n1w = P(); g.n2b = P(); g.n2w = P();
        return g;
    };
    Gcl a_inv[2]; a_inv[0] = getGcl(); a_inv[1] = getGcl();
    Gcl f_inv[2]; f_inv[0] = getGcl(); f_inv[1] = getGcl();

    // workspace carve (floats)
    float* w = (float*)d_ws;
    size_t off = 0;
    auto alloc = [&](size_t n) { float* p = w + off; off += n; return p; };
    float* haA     = alloc((size_t)NA * HH);
    float* haB     = alloc((size_t)NA * HH);
    float* hfA     = alloc((size_t)NF * HH);
    float* hfB     = alloc((size_t)NF * HH);
    float* efat2   = alloc((size_t)EF * 2);
    float* eaat2   = alloc((size_t)EA * 2);
    float* cda     = alloc((size_t)EA * 3);
    float* haagg   = alloc((size_t)NF * HH);
    float* hfext   = alloc((size_t)NA * HH);
    float* aggF    = alloc((size_t)NF * HH);
    float* aggA    = alloc((size_t)NA * HH);
    float* bmxf    = alloc((size_t)NA * 3);
    float* cdf     = alloc((size_t)NA * 3);
    float* radf    = alloc((size_t)NA);
    float* cAgg    = alloc((size_t)NA * 3);

    hipMemcpyAsync(haA, h_a_in, (size_t)NA * HH * sizeof(float), hipMemcpyDeviceToDevice, stream);
    hipMemcpyAsync(hfA, h_f_in, (size_t)NF * HH * sizeof(float), hipMemcpyDeviceToDevice, stream);

    c2d_kernel<<<(EF + 255) / 256, 256, 0, stream>>>(x_f, fr, fc, ef_attr, efat2, nullptr, EF);
    c2d_kernel<<<(EA + 255) / 256, 256, 0, stream>>>(x_a, ar, ac, ea_attr, eaat2, cda, EA);

    float* ha = haA; float* haN = haB;
    float* hf = hfA; float* hfN = hfB;
    const float scale = 0.01f;   // 1 / NORM_FACTOR

    for (int layer = 0; layer < 2; ++layer) {
        const Gcl& pf = f_inv[layer];
        const Gcl& pa = a_inv[layer];

        // fragment GCL
        spmm_kernel<<<NF, 256, 0, stream>>>(m_mat, ha, haagg, NA, HH, 1);
        fill_kernel<<<(int)(((size_t)NF * HH + 255) / 256), 256, 0, stream>>>(aggF, (size_t)NF * HH);
        edge_mlp_kernel<0><<<(EF + 15) / 16, 256, 0, stream>>>(
            hf, fr, fc, efat2, nullptr, nullptr,
            pf.e1w, pf.e1b, pf.e2w, pf.e2b, pf.attw, pf.attb,
            nullptr, aggF, EF, 2 * HH + 2, scale);
        node_gcl_kernel<<<(NF + 15) / 16, 256, 0, stream>>>(
            hf, haagg, aggF, pf.n1w, pf.n1b, pf.n2w, pf.n2b, hfN, NF);
        { float* t = hf; hf = hfN; hfN = t; }

        // atom GCL
        spmm_kernel<<<NA, 256, 0, stream>>>(bm_mat, hf, hfext, NF, HH, 0);
        fill_kernel<<<(int)(((size_t)NA * HH + 255) / 256), 256, 0, stream>>>(aggA, (size_t)NA * HH);
        edge_mlp_kernel<0><<<(EA + 15) / 16, 256, 0, stream>>>(
            ha, ar, ac, eaat2, ea_type, pa.bond,
            pa.e1w, pa.e1b, pa.e2w, pa.e2b, pa.attw, pa.attb,
            nullptr, aggA, EA, 3 * HH + 2, scale);
        node_gcl_kernel<<<(NA + 15) / 16, 256, 0, stream>>>(
            ha, hfext, aggA, pa.n1w, pa.n1b, pa.n2w, pa.n2b, haN, NA);
        { float* t = ha; ha = haN; haN = t; }
    }

    // Equivariant update
    spmm_kernel<<<NA, 256, 0, stream>>>(bm_mat, x_f, bmxf, NF, 3, 0);
    cdf_kernel<<<(NA + 255) / 256, 256, 0, stream>>>(x_a, bmxf, cdf, radf, NA);
    spmm_kernel<<<NA, 256, 0, stream>>>(bm_mat, hf, hfext, NF, HH, 0);
    fill_kernel<<<(int)(((size_t)NA * 3 + 255) / 256), 256, 0, stream>>>(cAgg, (size_t)NA * 3);

    edge_mlp_kernel<1><<<(EA + 15) / 16, 256, 0, stream>>>(
        ha, ar, ac, eaat2, ea_type, eq_bond,
        eq_atom1_w, eq_atom1_b, eq_atom2_w, eq_atom2_b, eq_atom3_w, nullptr,
        cda, cAgg, EA, 3 * HH + 2, scale);

    float* outHa    = (float*)d_out;
    float* outHf    = outHa + (size_t)NA * HH;
    float* outCoord = outHf + (size_t)NF * HH;

    node_eq_kernel<<<(NA + 15) / 16, 256, 0, stream>>>(
        ha, hfext, radf,
        eq_frag1_w, eq_frag1_b, eq_frag2_w, eq_frag2_b, eq_frag3_w,
        cdf, x_a, cAgg, outCoord, NA);

    hipMemcpyAsync(outHa, ha, (size_t)NA * HH * sizeof(float), hipMemcpyDeviceToDevice, stream);
    hipMemcpyAsync(outHf, hf, (size_t)NF * HH * sizeof(float), hipMemcpyDeviceToDevice, stream);
}
